// FlashAttention_26070451487309
// MI455X (gfx1250) — compile-verified, matrix operand order checked by reading, not run on hardware
//
#include <hip/hip_runtime.h>
#include <hip/hip_bf16.h>
#include <stdint.h>

#define B_  2
#define S_  4096
#define DM  1024
#define H_  16
#define HD  64

typedef __attribute__((ext_vector_type(16))) _Float16 v16h;
typedef __attribute__((ext_vector_type(8)))  _Float16 v8h;
typedef __attribute__((ext_vector_type(8)))  float    v8f;
typedef __attribute__((ext_vector_type(4)))  int      v4i;

union Frag16 { v16h v; v8h h[2]; };

// ---------------------------------------------------------------------------
// Async global->LDS 128-bit copy (ASYNCcnt path); sync fallback if the
// builtin is unavailable. The builtin wants v4i pointers in as(1)/as(3);
// LDS offset = low 32 bits of the generic address per ISA 10.2.
// ---------------------------------------------------------------------------
__device__ __forceinline__ void async_cp16(const _Float16* g, _Float16* l) {
#if defined(__gfx1250__) && __has_builtin(__builtin_amdgcn_global_load_async_to_lds_b128)
  __builtin_amdgcn_global_load_async_to_lds_b128(
      (__attribute__((address_space(1))) v4i*)(unsigned long long)(uintptr_t)g,
      (__attribute__((address_space(3))) v4i*)(unsigned int)(uintptr_t)l,
      0, 0);
#else
  *(v8h*)l = *(const v8h*)g;
#endif
}

__device__ __forceinline__ void wait_async() {
#if defined(__gfx1250__) && __has_builtin(__builtin_amdgcn_s_wait_asynccnt)
  __builtin_amdgcn_s_wait_asynccnt(0);
#elif defined(__gfx1250__)
  asm volatile("s_wait_asynccnt 0" ::: "memory");
#endif
}

// Per ISA 7.12.2 (16-bit A 16x32): lane -> row = lane%16, K-half = lane/16;
// element e in [0,7]  -> K = kOff + half*8 + e
// element e in [8,15] -> K = kOff + 16 + half*8 + (e-8)
__device__ __forceinline__ v16h lds_frag(const _Float16* p, int row, int ldm,
                                         int kOff, int half) {
  Frag16 f;
  f.h[0] = *(const v8h*)(p + row * ldm + kOff + half * 8);
  f.h[1] = *(const v8h*)(p + row * ldm + kOff + 16 + half * 8);
  return f.v;
}

__device__ __forceinline__ v8f wmma16(v16h a, v16h b, v8f c) {
  return __builtin_amdgcn_wmma_f32_16x16x32_f16(false, a, false, b,
                                                (short)0, c, false, false);
}

// ---------------------------------------------------------------------------
// One-time f32 -> f16 conversion (x and the four weight matrices).
// ---------------------------------------------------------------------------
__global__ __launch_bounds__(256)
void cvt_f32_f16_kernel(const float* __restrict__ s, _Float16* __restrict__ d,
                        int n8) {
  int i = blockIdx.x * 256 + threadIdx.x;
  if (i < n8) {
    const float4 a = *(const float4*)(s + (size_t)i * 8);
    const float4 b = *(const float4*)(s + (size_t)i * 8 + 4);
    v8h o = {(_Float16)a.x, (_Float16)a.y, (_Float16)a.z, (_Float16)a.w,
             (_Float16)b.x, (_Float16)b.y, (_Float16)b.z, (_Float16)b.w};
    *(v8h*)(d + (size_t)i * 8) = o;
  }
}

// ---------------------------------------------------------------------------
// Stage one 128x32 A tile (async DMA) + 32x128 W tile transposed to [n][k].
// ---------------------------------------------------------------------------
__device__ __forceinline__ void gemm_stage(const _Float16* __restrict__ A,
                                           const _Float16* __restrict__ Wh,
                                           _Float16* aL, _Float16* bL,
                                           int bm, int bn, int k0, int tid) {
  #pragma unroll
  for (int p = 0; p < 2; ++p) {                  // 512 chunks of 8 halves
    int idx = p * 256 + tid;
    int row = idx >> 2;
    int c8  = (idx & 3) * 8;
    async_cp16(A + (size_t)(bm + row) * DM + k0 + c8, aL + row * 32 + c8);
  }
  #pragma unroll
  for (int p = 0; p < 2; ++p) {                  // 32 k-rows x 16 chunks of 8 n
    int idx = p * 256 + tid;
    int k  = idx >> 4;
    int n8 = (idx & 15) * 8;
    v8h w = *(const v8h*)(Wh + (size_t)(k0 + k) * DM + bn + n8);
    #pragma unroll
    for (int i = 0; i < 8; ++i) bL[(n8 + i) * 32 + k] = w[i];
  }
}

// ---------------------------------------------------------------------------
// Tiled GEMM: C[M=8192,1024] = A[M,1024](f16) * W[1024,1024](f16) (+bias)(*scale)
// OUT_BHSD: scatter f16 into [B,H,S,64]; else fp32 [M,1024] row-major.
// 256 thr = 8 waves (4 M-strips x 2 N-strips), tile 128x128, BK=32,
// double-buffered LDS with async-DMA A staging.
// ---------------------------------------------------------------------------
template <bool OUT_BHSD>
__global__ __launch_bounds__(256)
void gemm_wmma_kernel(const _Float16* __restrict__ A,
                      const _Float16* __restrict__ Wh,
                      const float* __restrict__ bias,
                      void* __restrict__ Out, float scale) {
  __shared__ __align__(16) _Float16 aLds[2][128 * 32];   // [row][k]
  __shared__ __align__(16) _Float16 bLds[2][128 * 32];   // W^T tile: [n][k]

  const int tid   = threadIdx.x;
  const int lane  = tid & 31;
  const int wave  = tid >> 5;
  const int waveM = wave & 3;
  const int waveN = wave >> 2;
  const int half  = lane >> 4;
  const int l16   = lane & 15;
  const int bm = blockIdx.x * 128;
  const int bn = blockIdx.y * 128;

  v8f acc[2][4];
  const v8f vzero = {0.f, 0.f, 0.f, 0.f, 0.f, 0.f, 0.f, 0.f};
  for (int mi = 0; mi < 2; ++mi)
    for (int j = 0; j < 4; ++j) acc[mi][j] = vzero;

  gemm_stage(A, Wh, aLds[0], bLds[0], bm, bn, 0, tid);
  wait_async();
  __syncthreads();

  const int NIT = DM / 32;
  for (int it = 0; it < NIT; ++it) {
    const int cur = it & 1;
    if (it + 1 < NIT)
      gemm_stage(A, Wh, aLds[cur ^ 1], bLds[cur ^ 1], bm, bn, (it + 1) * 32, tid);

    v16h afr[2], bfr[4];
    #pragma unroll
    for (int mi = 0; mi < 2; ++mi)
      afr[mi] = lds_frag(aLds[cur], waveM * 32 + mi * 16 + l16, 32, 0, half);
    #pragma unroll
    for (int j = 0; j < 4; ++j)
      bfr[j] = lds_frag(bLds[cur], waveN * 64 + j * 16 + l16, 32, 0, half);
    #pragma unroll
    for (int mi = 0; mi < 2; ++mi)
      #pragma unroll
      for (int j = 0; j < 4; ++j)
        acc[mi][j] = wmma16(afr[mi], bfr[j], acc[mi][j]);

    wait_async();
    __syncthreads();
  }

  // ---- epilogue: bias + scale, store (C layout: row = r + 8*half) ----
  #pragma unroll
  for (int mi = 0; mi < 2; ++mi)
    #pragma unroll
    for (int j = 0; j < 4; ++j) {
      int col = bn + waveN * 64 + j * 16 + l16;
      float bv = bias[col];
      #pragma unroll
      for (int r = 0; r < 8; ++r) {
        int row = bm + waveM * 32 + mi * 16 + r + 8 * half;
        float val = (acc[mi][j][r] + bv) * scale;
        if (OUT_BHSD) {
          int b = row >> 12, s = row & (S_ - 1);
          int h = col >> 6,  d = col & (HD - 1);
          ((_Float16*)Out)[(((size_t)(b * H_ + h)) * S_ + s) * HD + d] =
              (_Float16)val;
        } else {
          ((float*)Out)[(size_t)row * DM + col] = val;
        }
      }
    }
}

// ---------------------------------------------------------------------------
// Stage one 64-key KV block: K via async DMA, V transposed to [d][key].
// ---------------------------------------------------------------------------
__device__ __forceinline__ void attn_stage_kv(const _Float16* __restrict__ kg,
                                              const _Float16* __restrict__ vg,
                                              _Float16* kL, _Float16* vTL,
                                              int tid) {
  #pragma unroll
  for (int p = 0; p < 4; ++p) {
    int idx = p * 128 + tid;                     // 512 chunks of 8 halves
    async_cp16(kg + (size_t)idx * 8, kL + idx * 8);
  }
  #pragma unroll
  for (int p = 0; p < 4; ++p) {
    int idx = p * 128 + tid;
    int key = idx >> 3;
    int d0  = (idx & 7) * 8;
    v8h d = *(const v8h*)(vg + (size_t)idx * 8);
    #pragma unroll
    for (int i = 0; i < 8; ++i) vTL[(d0 + i) * 64 + key] = d[i];
  }
}

// ---------------------------------------------------------------------------
// Flash attention: block = (64-query tile, b*H+h); 128 thr = 4 waves, each
// wave owns 16 query rows; KV in double-buffered 64-key blocks; Q pre-scaled.
// ---------------------------------------------------------------------------
__global__ __launch_bounds__(128)
void flash_attn_kernel(const _Float16* __restrict__ Q,
                       const _Float16* __restrict__ K,
                       const _Float16* __restrict__ V,
                       _Float16* __restrict__ Outf16) {
  __shared__ __align__(16) _Float16 qLds[64 * 64];      // [q][d]
  __shared__ __align__(16) _Float16 kLds[2][64 * 64];   // [key][d]
  __shared__ __align__(16) _Float16 vT  [2][64 * 64];   // [d][key]
  __shared__ __align__(16) _Float16 pLds[64 * 64];      // [q][key], wave-private

  const int tid  = threadIdx.x;
  const int lane = tid & 31;
  const int wave = tid >> 5;
  const int half = lane >> 4;
  const int l16  = lane & 15;
  const int bh = blockIdx.y;
  const int qt = blockIdx.x;

  const size_t qbase  = ((size_t)bh * S_ + (size_t)qt * 64) * HD;
  const size_t kvbase = (size_t)bh * S_ * HD;

  #pragma unroll
  for (int p = 0; p < 4; ++p) {
    int idx = p * 128 + tid;
    async_cp16(Q + qbase + (size_t)idx * 8, qLds + idx * 8);
  }
  attn_stage_kv(K + kvbase, V + kvbase, kLds[0], vT[0], tid);
  wait_async();
  __syncthreads();

  float m_i[8], l_i[8];
  v8f o[4];
  const v8f vzero = {0.f, 0.f, 0.f, 0.f, 0.f, 0.f, 0.f, 0.f};
  #pragma unroll
  for (int r = 0; r < 8; ++r) { m_i[r] = -1e30f; l_i[r] = 0.f; }
  #pragma unroll
  for (int j = 0; j < 4; ++j) o[j] = vzero;

  const int NB = S_ / 64;
  for (int blk = 0; blk < NB; ++blk) {
    const int cur = blk & 1;
    if (blk + 1 < NB)
      attn_stage_kv(K + kvbase + (size_t)(blk + 1) * 64 * HD,
                    V + kvbase + (size_t)(blk + 1) * 64 * HD,
                    kLds[cur ^ 1], vT[cur ^ 1], tid);
    if (blk + 2 < NB)   // L2 warm-up two blocks ahead
      __builtin_prefetch(
          (const void*)(K + kvbase + (size_t)(blk + 2) * 64 * HD + tid * 32),
          0, 1);

    // ---- scores: S = Q(16x64) * K^T(64x64) ----
    v8f sc[4];
    #pragma unroll
    for (int j = 0; j < 4; ++j) sc[j] = vzero;
    #pragma unroll
    for (int kk = 0; kk < 64; kk += 32) {
      v16h aq = lds_frag(qLds, wave * 16 + l16, 64, kk, half);
      #pragma unroll
      for (int j = 0; j < 4; ++j) {
        v16h bk = lds_frag(kLds[cur], j * 16 + l16, 64, kk, half);
        sc[j] = wmma16(aq, bk, sc[j]);
      }
    }

    // ---- online softmax (row = r + 8*half per C layout) ----
    float newm[8], alpha[8], rowsum[8];
    #pragma unroll
    for (int r = 0; r < 8; ++r) {
      float v = fmaxf(fmaxf(sc[0][r], sc[1][r]), fmaxf(sc[2][r], sc[3][r]));
      #pragma unroll
      for (int m = 1; m <= 8; m <<= 1) v = fmaxf(v, __shfl_xor(v, m, 32));
      newm[r]  = fmaxf(m_i[r], v);
      alpha[r] = __expf(m_i[r] - newm[r]);
      rowsum[r] = 0.f;
    }
    #pragma unroll
    for (int j = 0; j < 4; ++j)
      #pragma unroll
      for (int r = 0; r < 8; ++r) {
        float p = __expf(sc[j][r] - newm[r]);
        sc[j][r] = p;
        rowsum[r] += p;
      }
    #pragma unroll
    for (int r = 0; r < 8; ++r) {
      float v = rowsum[r];
      #pragma unroll
      for (int m = 1; m <= 8; m <<= 1) v += __shfl_xor(v, m, 32);
      l_i[r] = alpha[r] * l_i[r] + v;
      m_i[r] = newm[r];
    }

    // ---- P to LDS f16 (wave-private rows), rescale O ----
    #pragma unroll
    for (int j = 0; j < 4; ++j)
      #pragma unroll
      for (int r = 0; r < 8; ++r)
        pLds[(wave * 16 + r + 8 * half) * 64 + j * 16 + l16] =
            (_Float16)sc[j][r];
    asm volatile("s_wait_dscnt 0" ::: "memory");   // wave-private LDS RAW
    #pragma unroll
    for (int j = 0; j < 4; ++j)
      #pragma unroll
      for (int r = 0; r < 8; ++r) o[j][r] *= alpha[r];

    // ---- O += P(16x64) * V(64x64) ----
    #pragma unroll
    for (int kk = 0; kk < 64; kk += 32) {
      v16h ap = lds_frag(pLds, wave * 16 + l16, 64, kk, half);
      #pragma unroll
      for (int j = 0; j < 4; ++j) {
        v16h bv = lds_frag(vT[cur], j * 16 + l16, 64, kk, half);
        o[j] = wmma16(ap, bv, o[j]);
      }
    }

    wait_async();
    __syncthreads();
  }

  // ---- normalize, store f16 to [B,S,DM] for the O-projection GEMM ----
  const int b = bh / H_;
  const int h = bh % H_;
  #pragma unroll
  for (int j = 0; j < 4; ++j)
    #pragma unroll
    for (int r = 0; r < 8; ++r) {
      int srow = qt * 64 + wave * 16 + r + 8 * half;
      int d    = j * 16 + l16;
      float val = o[j][r] / l_i[r];
      Outf16[((size_t)b * S_ + srow) * DM + h * HD + d] = (_Float16)val;
    }
}

// ---------------------------------------------------------------------------
extern "C" void kernel_launch(void* const* d_in, const int* in_sizes, int n_in,
                              void* d_out, int out_size, void* d_ws,
                              size_t ws_size, hipStream_t stream) {
  const float* x  = (const float*)d_in[0];
  const float* Wq = (const float*)d_in[1];
  const float* bq = (const float*)d_in[2];
  const float* Wk = (const float*)d_in[3];
  const float* bk = (const float*)d_in[4];
  const float* Wv = (const float*)d_in[5];
  const float* bv = (const float*)d_in[6];
  const float* Wo = (const float*)d_in[7];
  const float* bo = (const float*)d_in[8];

  const size_t elems = (size_t)B_ * S_ * DM;   // 8 Mi
  const size_t welem = (size_t)DM * DM;        // 1 Mi
  _Float16* xh  = (_Float16*)d_ws;             // f16 activations
  _Float16* wqh = xh + elems;
  _Float16* wkh = wqh + welem;
  _Float16* wvh = wkh + welem;
  _Float16* woh = wvh + welem;
  _Float16* qws = woh + welem;                 // [B,H,S,64], pre-scaled 1/8
  _Float16* kws = qws + elems;
  _Float16* vws = kws + elems;
  _Float16* aws = vws + elems;                 // attention out, [B,S,DM]

  // one-time f32 -> f16 conversions
  cvt_f32_f16_kernel<<<(int)(elems / 8 / 256), 256, 0, stream>>>(x, xh,
                                                                 (int)(elems / 8));
  cvt_f32_f16_kernel<<<(int)(welem / 8 / 256), 256, 0, stream>>>(Wq, wqh,
                                                                 (int)(welem / 8));
  cvt_f32_f16_kernel<<<(int)(welem / 8 / 256), 256, 0, stream>>>(Wk, wkh,
                                                                 (int)(welem / 8));
  cvt_f32_f16_kernel<<<(int)(welem / 8 / 256), 256, 0, stream>>>(Wv, wvh,
                                                                 (int)(welem / 8));
  cvt_f32_f16_kernel<<<(int)(welem / 8 / 256), 256, 0, stream>>>(Wo, woh,
                                                                 (int)(welem / 8));

  dim3 gGemm(64, 8);   // 8192/128 x 1024/128
  dim3 bGemm(256);
  gemm_wmma_kernel<true><<<gGemm, bGemm, 0, stream>>>(xh, wqh, bq, (void*)qws,
                                                      0.125f);  // 1/sqrt(64)
  gemm_wmma_kernel<true><<<gGemm, bGemm, 0, stream>>>(xh, wkh, bk, (void*)kws,
                                                      1.0f);
  gemm_wmma_kernel<true><<<gGemm, bGemm, 0, stream>>>(xh, wvh, bv, (void*)vws,
                                                      1.0f);

  dim3 gAttn(S_ / 64, B_ * H_);
  flash_attn_kernel<<<gAttn, dim3(128), 0, stream>>>(qws, kws, vws, aws);

  gemm_wmma_kernel<false><<<gGemm, bGemm, 0, stream>>>(aws, woh, bo, d_out,
                                                       1.0f);
}
